// MCPRN_10247791969040
// MI455X (gfx1250) — matrix operand order repaired and checked by reading, not compiled, not run on hardware
//
#include <hip/hip_runtime.h>
#include <math.h>

#define SEQL   50
#define BATCH  128
#define DIM    256
#define NPURP  3
#define NITEMS 50001
#define GDIM   768
#define TAU    0.1f
#define CEPS   0.01f

typedef __attribute__((ext_vector_type(2)))  float    v2f;
typedef __attribute__((ext_vector_type(8)))  float    v8f;
typedef __attribute__((ext_vector_type(16))) _Float16 v16h;
typedef __attribute__((ext_vector_type(8)))  _Float16 v8h;

union V16H { v16h v; v8h h2[2]; _Float16 e[16]; };

__device__ __forceinline__ float sigm(float x) { return 1.0f / (1.0f + expf(-x)); }

// ---- CDNA5 async memory->LDS staging (ASYNCcnt-tracked, no VGPR round trip) ----
__device__ __forceinline__ void async_copy16(unsigned int lds_addr, const void* gsrc) {
    asm volatile("global_load_async_to_lds_b128 %0, %1, off"
                 :: "v"(lds_addr), "v"(gsrc) : "memory");
}
__device__ __forceinline__ void async_wait0() {
    asm volatile("s_wait_asynccnt 0x0" ::: "memory");
}
__device__ __forceinline__ unsigned int lds_addr_of(const void* p) {
    return (unsigned int)(unsigned long long)p;   // generic shared ptr: addr[31:0] = LDS offset
}

// ---------------- utility kernels ----------------

__global__ void zero_f32(float* p, int n) {
    int i = blockIdx.x * 256 + threadIdx.x;
    if (i < n) p[i] = 0.0f;
}

__global__ void cvt_f32_f16(const float* __restrict__ src, _Float16* __restrict__ dst, int n) {
    int i = blockIdx.x * 256 + threadIdx.x;
    if (i < n) dst[i] = (_Float16)src[i];
}

// ---------------- gather + concentration softmax ----------------

__global__ void __launch_bounds__(256) gather_concen(
    const int* __restrict__ seq, const float* __restrict__ emb,
    const float* __restrict__ embp, float* __restrict__ X, float* __restrict__ concen) {
    const int b = blockIdx.x, s = blockIdx.y, tid = threadIdx.x;
    __shared__ float red[256];
    __shared__ float sc[NPURP];
    const int item = seq[s * BATCH + b];
    const float x = emb[(size_t)item * DIM + tid];
    X[((size_t)s * BATCH + b) * DIM + tid] = x;
    for (int p = 0; p < NPURP; p++) {
        red[tid] = x * embp[p * DIM + tid];
        __syncthreads();
        for (int off = 128; off > 0; off >>= 1) {
            if (tid < off) red[tid] += red[tid + off];
            __syncthreads();
        }
        if (tid == 0) sc[p] = red[0];
        __syncthreads();
    }
    if (tid == 0) {
        float m  = fmaxf(sc[0], fmaxf(sc[1], sc[2]));
        float e0 = expf((sc[0] - m) / TAU);
        float e1 = expf((sc[1] - m) / TAU);
        float e2 = expf((sc[2] - m) / TAU);
        float inv = 1.0f / (e0 + e1 + e2);
        float mask = (item != 0) ? 1.0f : 0.0f;
        float* c = concen + ((size_t)s * BATCH + b) * NPURP;
        c[0] = e0 * inv * mask; c[1] = e1 * inv * mask; c[2] = e2 * inv * mask;
    }
}

// ---------------- one PSRU time step (fp32 WMMA 16x16x4) ----------------
// grid: (BATCH/16, DIM/64, NPURP), block: 128 threads (4 waves)

__global__ void __launch_bounds__(128) psru_step(
    int s, const float* __restrict__ X, const float* __restrict__ concen,
    const float* __restrict__ w_ih, const float* __restrict__ w_hh,
    const float* __restrict__ b_ih, const float* __restrict__ b_hh,
    const float* __restrict__ hin, float* __restrict__ hout) {
    const int b0   = blockIdx.x * 16;
    const int cgrp = blockIdx.y;
    const int p    = blockIdx.z;
    const int tid  = threadIdx.x;

    __shared__ float xa[16][DIM + 4];
    __shared__ float ha[16][DIM + 4];

    // async-stage both 16x256 fp32 tiles (16B chunks; row stride (DIM+4)*4 is 16B aligned)
    for (int c = tid; c < 16 * (DIM / 4); c += 128) {
        const int m  = c >> 6;             // 64 16B-chunks per row
        const int k4 = (c & 63) * 4;
        async_copy16(lds_addr_of(&xa[m][k4]),
                     X + ((size_t)s * BATCH + b0 + m) * DIM + k4);
        async_copy16(lds_addr_of(&ha[m][k4]),
                     hin + ((size_t)p * BATCH + b0 + m) * DIM + k4);
    }
    async_wait0();
    __syncthreads();

    const int wave = tid >> 5, lane = tid & 31;
    const int g = lane >> 4, ln = lane & 15;
    const int jc = cgrp * 64 + wave * 16 + ln;

    const float* wi = w_ih + (size_t)p * GDIM * DIM;
    const float* wh = w_hh + (size_t)p * GDIM * DIM;
    const float* rowi_r = wi + (size_t)(jc      ) * DIM;
    const float* rowi_i = wi + (size_t)(jc + 256) * DIM;
    const float* rowi_n = wi + (size_t)(jc + 512) * DIM;
    const float* rowh_r = wh + (size_t)(jc      ) * DIM;
    const float* rowh_i = wh + (size_t)(jc + 256) * DIM;
    const float* rowh_n = wh + (size_t)(jc + 512) * DIM;

    v8f a_ir = {0}, a_hr = {0}, a_ii = {0}, a_hi = {0}, a_in = {0}, a_hn = {0};

    for (int kc = 0; kc < DIM; kc += 4) {
        const int ko = kc + 2 * g;                  // A/B: VGPR v -> K = v + 2*(lane/16)
        v2f ax  = *(const v2f*)&xa[ln][ko];
        v2f ah  = *(const v2f*)&ha[ln][ko];
        v2f bir = *(const v2f*)(rowi_r + ko);
        v2f bhr = *(const v2f*)(rowh_r + ko);
        v2f bii = *(const v2f*)(rowi_i + ko);
        v2f bhi = *(const v2f*)(rowh_i + ko);
        v2f bin = *(const v2f*)(rowi_n + ko);
        v2f bhn = *(const v2f*)(rowh_n + ko);
        a_ir = __builtin_amdgcn_wmma_f32_16x16x4_f32(false, ax, false, bir, (short)0, a_ir, false, false);
        a_hr = __builtin_amdgcn_wmma_f32_16x16x4_f32(false, ah, false, bhr, (short)0, a_hr, false, false);
        a_ii = __builtin_amdgcn_wmma_f32_16x16x4_f32(false, ax, false, bii, (short)0, a_ii, false, false);
        a_hi = __builtin_amdgcn_wmma_f32_16x16x4_f32(false, ah, false, bhi, (short)0, a_hi, false, false);
        a_in = __builtin_amdgcn_wmma_f32_16x16x4_f32(false, ax, false, bin, (short)0, a_in, false, false);
        a_hn = __builtin_amdgcn_wmma_f32_16x16x4_f32(false, ah, false, bhn, (short)0, a_hn, false, false);
    }

    const float bir_s = b_ih[p * GDIM + jc],       bhr_s = b_hh[p * GDIM + jc];
    const float bii_s = b_ih[p * GDIM + 256 + jc], bhi_s = b_hh[p * GDIM + 256 + jc];
    const float bin_s = b_ih[p * GDIM + 512 + jc], bhn_s = b_hh[p * GDIM + 512 + jc];

#pragma unroll
    for (int r = 0; r < 8; r++) {
        const int m = r + 8 * g;                    // D: VGPR r -> M = r + 8*(lane/16)
        float ir  = a_ir[r] + bir_s, hr  = a_hr[r] + bhr_s;
        float ii  = a_ii[r] + bii_s, hii = a_hi[r] + bhi_s;
        float inn = a_in[r] + bin_s, hnn = a_hn[r] + bhn_s;
        float rg = sigm(ir + hr);
        float ig = sigm(ii + hii);
        float ng = tanhf(inn + rg * hnn);
        float hold = ha[m][jc];
        float c  = concen[((size_t)s * BATCH + b0 + m) * NPURP + p];
        float cf = (c >= CEPS) ? c : 0.0f;
        hout[((size_t)p * BATCH + b0 + m) * DIM + jc] = hold - cf * ig * (hold - ng);
    }
}

// ---------------- scoring (f16 WMMA 16x16x32, fp32 accumulate) ----------------
// grid: ceil(NITEMS/32) blocks, 256 threads (8 waves = all 128 batch columns).
// M-tile = 32 rows (2 sub-tiles) per block; hn pre-converted to f16.

__global__ void __launch_bounds__(256) score_kernel(
    const _Float16* __restrict__ embh, const float* __restrict__ embp,
    const _Float16* __restrict__ hnh, float* __restrict__ out) {
    const int t0  = blockIdx.x * 32;
    const int tid = threadIdx.x;
    __shared__ _Float16 eah[32][DIM + 8];
    __shared__ float psc[32][NPURP];
    __shared__ float tcw_s[32][NPURP];

    // async-stage 32x256 f16 tile; OOB rows clamp to the all-zero padding row 0
    for (int c = tid; c < 32 * (DIM / 8); c += 256) {
        const int m   = c >> 5;            // 32 16B-chunks per row
        const int seg = (c & 31) * 8;
        const int t   = t0 + m;
        const int tr  = (t < NITEMS) ? t : 0;
        async_copy16(lds_addr_of(&eah[m][seg]), embh + (size_t)tr * DIM + seg);
    }
    async_wait0();
    __syncthreads();

    if (tid < 32 * NPURP) {                // tcw raw scores
        const int m = tid / NPURP, pp = tid % NPURP;
        const float* ep = embp + pp * DIM;
        float d = 0.0f;
        for (int k = 0; k < DIM; k++) d += (float)eah[m][k] * ep[k];
        psc[m][pp] = d;
    }
    __syncthreads();
    if (tid < 32) {                        // softmax over purposes (no tau)
        float a = psc[tid][0], b = psc[tid][1], c = psc[tid][2];
        float mx = fmaxf(a, fmaxf(b, c));
        float e0 = expf(a - mx), e1 = expf(b - mx), e2 = expf(c - mx);
        float inv = 1.0f / (e0 + e1 + e2);
        tcw_s[tid][0] = e0 * inv; tcw_s[tid][1] = e1 * inv; tcw_s[tid][2] = e2 * inv;
    }
    __syncthreads();

    const int wave = tid >> 5, lane = tid & 31;
    const int g = lane >> 4, ln = lane & 15;
    const int b = wave * 16 + ln;          // batch column (B fragment N lane)

    // prefetch next block's tile (emits global_prefetch_b8)
    if ((size_t)(t0 + 32) * DIM + tid * 16 < (size_t)NITEMS * DIM)
        __builtin_prefetch(embh + (size_t)(t0 + 32) * DIM + tid * 16, 0, 1);

    v8f acc[2][NPURP] = {};
    for (int kc = 0; kc < DIM; kc += 32) {
        V16H a0, a1;                       // A: idx<8 -> K=kc+idx+8g ; idx>=8 -> K=kc+16+(idx-8)+8g
        a0.h2[0] = *(const v8h*)&eah[ln][kc + 8 * g];
        a0.h2[1] = *(const v8h*)&eah[ln][kc + 16 + 8 * g];
        a1.h2[0] = *(const v8h*)&eah[16 + ln][kc + 8 * g];
        a1.h2[1] = *(const v8h*)&eah[16 + ln][kc + 16 + 8 * g];
        const int kb = kc + 8 * g;         // B: idx<8 -> K=kc+idx+8g ; idx>=8 -> K=kc+16+(idx-8)+8g
#pragma unroll
        for (int pp = 0; pp < NPURP; pp++) {
            const _Float16* hrow = hnh + ((size_t)pp * BATCH + b) * DIM + kb;
            V16H bf;
            bf.h2[0] = *(const v8h*)(hrow);
            bf.h2[1] = *(const v8h*)(hrow + 16);
            acc[0][pp] = __builtin_amdgcn_wmma_f32_16x16x32_f16(
                false, a0.v, false, bf.v, (short)0, acc[0][pp], false, false);
            acc[1][pp] = __builtin_amdgcn_wmma_f32_16x16x32_f16(
                false, a1.v, false, bf.v, (short)0, acc[1][pp], false, false);
        }
    }

#pragma unroll
    for (int u = 0; u < 2; u++) {
#pragma unroll
        for (int r = 0; r < 8; r++) {
            const int m = u * 16 + r + 8 * g;
            const int t = t0 + m;
            if (t < NITEMS) {
                float v = tcw_s[m][0] * acc[u][0][r] + tcw_s[m][1] * acc[u][1][r]
                        + tcw_s[m][2] * acc[u][2][r];
                out[(size_t)b * NITEMS + t] = v;
            }
        }
    }
}

// ---------------- host side ----------------

extern "C" void kernel_launch(void* const* d_in, const int* in_sizes, int n_in,
                              void* d_out, int out_size, void* d_ws, size_t ws_size,
                              hipStream_t stream) {
    (void)in_sizes; (void)n_in; (void)out_size; (void)ws_size;
    const int*   seq  = (const int*)d_in[0];
    const float* emb  = (const float*)d_in[1];
    const float* embp = (const float*)d_in[2];
    const float* w_ih = (const float*)d_in[3];
    const float* w_hh = (const float*)d_in[4];
    const float* b_ih = (const float*)d_in[5];
    const float* b_hh = (const float*)d_in[6];
    float* out = (float*)d_out;

    float* X      = (float*)d_ws;                         // [S][B][D] f32
    float* concen = X + (size_t)SEQL * BATCH * DIM;       // [S][B][P] f32
    float* h0     = concen + (size_t)SEQL * BATCH * NPURP;
    float* h1     = h0 + (size_t)NPURP * BATCH * DIM;
    _Float16* embh = (_Float16*)(h1 + (size_t)NPURP * BATCH * DIM);  // [T][D] f16
    _Float16* hnh  = embh + (size_t)NITEMS * DIM;                    // [P][B][D] f16

    {
        int n = NPURP * BATCH * DIM;
        zero_f32<<<(n + 255) / 256, 256, 0, stream>>>(h0, n);
    }
    {
        int n = NITEMS * DIM;
        cvt_f32_f16<<<(n + 255) / 256, 256, 0, stream>>>(emb, embh, n);
    }
    gather_concen<<<dim3(BATCH, SEQL), 256, 0, stream>>>(seq, emb, embp, X, concen);

    float* hb[2] = {h0, h1};
    for (int s = 0; s < SEQL; s++) {
        psru_step<<<dim3(BATCH / 16, DIM / 64, NPURP), 128, 0, stream>>>(
            s, X, concen, w_ih, w_hh, b_ih, b_hh, hb[s & 1], hb[(s + 1) & 1]);
    }
    {
        int n = NPURP * BATCH * DIM;
        cvt_f32_f16<<<(n + 255) / 256, 256, 0, stream>>>(hb[SEQL & 1], hnh, n);
    }
    score_kernel<<<(NITEMS + 31) / 32, 256, 0, stream>>>(embh, embp, hnh, out);
}
